// MultiHeadAttention_31980326486208
// MI455X (gfx1250) — matrix operand order repaired and checked by reading, not compiled, not run on hardware
//
#include <hip/hip_runtime.h>
#include <hip/hip_bf16.h>

typedef __attribute__((ext_vector_type(16))) __bf16 bf16x16;
typedef __attribute__((ext_vector_type(8)))  __bf16 bf16x8;
typedef __attribute__((ext_vector_type(8)))  float  f32x8;

// LDS byte offset of a __shared__ pointer: per CDNA5 aperture rules the LDS
// address is the low 32 bits of the flat address.
template <typename T>
__device__ __forceinline__ uint32_t lds_off(const T* p) {
    return (uint32_t)(uintptr_t)p;
}

// Async memory->LDS copy of 16 bytes per lane (GVS mode), tracked by ASYNCcnt.
__device__ __forceinline__ void async_b128(uint32_t lds_addr, uint32_t goff, uint64_t sbase) {
    asm volatile("global_load_async_to_lds_b128 %0, %1, %2"
                 :: "v"(lds_addr), "v"(goff), "s"(sbase) : "memory");
}
__device__ __forceinline__ void wait_async0() {
    asm volatile("s_wait_asynccnt 0" ::: "memory");
}

// ---------------------------------------------------------------------------
// Kernel 0: f32 -> bf16 conversion (weights)
// ---------------------------------------------------------------------------
__global__ __launch_bounds__(256) void cvt_f32_bf16_kernel(const float* __restrict__ src,
                                                           __bf16* __restrict__ dst, int n) {
    int i = (blockIdx.x * 256 + threadIdx.x) * 4;
    if (i + 3 < n) {
        float4 f = *(const float4*)(src + i);
        dst[i + 0] = (__bf16)f.x;
        dst[i + 1] = (__bf16)f.y;
        dst[i + 2] = (__bf16)f.z;
        dst[i + 3] = (__bf16)f.w;
    }
}

// ---------------------------------------------------------------------------
// GEMM:  Out[M,N] = A[M,K] @ W[N,K]^T + bias[N]
// Block tile 128(M) x 64(N), K-step 64 (two WMMA k-steps per stage -> 8 WMMA
// per barrier pair per wave). 8 waves, wave tile 32x32 (4 f32 accumulators).
// W (and bf16 A) staged via global_load_async_to_lds_b128; f32 A converted
// to bf16 through VGPRs.
// ---------------------------------------------------------------------------
template <bool AF32, bool OUTF32>
__global__ __launch_bounds__(256) void gemm_wmma_kernel(const void* __restrict__ Ap,
                                                        const __bf16* __restrict__ W,
                                                        const float* __restrict__ bias,
                                                        void* __restrict__ Op,
                                                        int M, int N, int K) {
    constexpr int BK = 64;
    constexpr int AS = 72;  // row stride (64 + 8 pad) -> 144B, 16B aligned
    constexpr int BS = 72;
    __shared__ __attribute__((aligned(16))) __bf16 As[128 * AS];
    __shared__ __attribute__((aligned(16))) __bf16 Bs[64 * BS];

    const int t    = threadIdx.x;
    const int lane = t & 31;
    const int wave = t >> 5;
    const int hi   = lane >> 4;
    const int l15  = lane & 15;
    const int wm   = wave >> 1;  // 0..3  (M direction, 32 rows each)
    const int wn   = wave & 1;   // 0..1  (N direction, 32 cols each)

    const int n0 = blockIdx.x * 64;
    const int m0 = blockIdx.y * 128;

    f32x8 acc[2][2];
#pragma unroll
    for (int mi = 0; mi < 2; ++mi)
#pragma unroll
        for (int ni = 0; ni < 2; ++ni)
#pragma unroll
            for (int r = 0; r < 8; ++r) acc[mi][ni][r] = 0.0f;

    const float*  Af = (const float*)Ap;
    const __bf16* Ab = (const __bf16*)Ap;

    for (int kk = 0; kk < K; kk += BK) {
        __syncthreads();  // protect LDS reuse from previous iteration
        {
            // --- stage A tile: 128 rows x 64 cols (32 elems per thread) ---
            const int r  = t >> 1;
            const int c0 = (t & 1) * 32;
            if constexpr (AF32) {
                const float4* s4 = (const float4*)(Af + (size_t)(m0 + r) * K + kk + c0);
#pragma unroll
                for (int j = 0; j < 4; ++j) {
                    float4 f0 = s4[2 * j], f1 = s4[2 * j + 1];
                    bf16x8 u = {(__bf16)f0.x, (__bf16)f0.y, (__bf16)f0.z, (__bf16)f0.w,
                                (__bf16)f1.x, (__bf16)f1.y, (__bf16)f1.z, (__bf16)f1.w};
                    *(bf16x8*)&As[r * AS + c0 + 8 * j] = u;
                }
                if (kk + BK < K)  // cover next tile's HBM latency
                    __builtin_prefetch(Af + (size_t)(m0 + r) * K + kk + BK + c0, 0, 0);
            } else {
                const uint32_t lds = lds_off(&As[r * AS + c0]);
                const uint32_t go  = (uint32_t)(((size_t)(m0 + r) * K + kk + c0) * 2);
#pragma unroll
                for (int j = 0; j < 4; ++j)
                    async_b128(lds + 16u * j, go + 16u * j, (uint64_t)Ab);
            }
            // --- stage B tile: W rows n0..n0+63, cols kk..kk+63, stored [n][k] ---
            const int n  = t >> 2;
            const int c1 = (t & 3) * 16;
            const uint32_t ldsb = lds_off(&Bs[n * BS + c1]);
            const uint32_t gob  = (uint32_t)(((size_t)(n0 + n) * K + kk + c1) * 2);
#pragma unroll
            for (int j = 0; j < 2; ++j)
                async_b128(ldsb + 16u * j, gob + 16u * j, (uint64_t)W);
        }
        wait_async0();
        __syncthreads();

        // --- two WMMA k-steps per staged tile ---
#pragma unroll
        for (int kt = 0; kt < 2; ++kt) {
            bf16x16 afr[2], bfr[2];
#pragma unroll
            for (int mi = 0; mi < 2; ++mi) {
                const int row = wm * 32 + mi * 16 + l15;
                bf16x8 lo = *(const bf16x8*)&As[row * AS + kt * 32 + 8 * hi];
                bf16x8 h8 = *(const bf16x8*)&As[row * AS + kt * 32 + 16 + 8 * hi];
#pragma unroll
                for (int i = 0; i < 8; ++i) { afr[mi][i] = lo[i]; afr[mi][i + 8] = h8[i]; }
            }
#pragma unroll
            for (int ni = 0; ni < 2; ++ni) {
                const int col = wn * 32 + ni * 16 + l15;
                bf16x8 lo = *(const bf16x8*)&Bs[col * BS + kt * 32 + 8 * hi];
                bf16x8 h8 = *(const bf16x8*)&Bs[col * BS + kt * 32 + 16 + 8 * hi];
#pragma unroll
                for (int i = 0; i < 8; ++i) { bfr[ni][i] = lo[i]; bfr[ni][i + 8] = h8[i]; }
            }
#pragma unroll
            for (int mi = 0; mi < 2; ++mi)
#pragma unroll
                for (int ni = 0; ni < 2; ++ni)
                    acc[mi][ni] = __builtin_amdgcn_wmma_f32_16x16x32_bf16(
                        false, afr[mi], false, bfr[ni], (short)0, acc[mi][ni], false, false);
        }
    }

    // --- epilogue: bias add + store (C/D layout: VGPR r -> row r + 8*hi, col lane%16) ---
    float*  Of = (float*)Op;
    __bf16* Ob = (__bf16*)Op;
#pragma unroll
    for (int mi = 0; mi < 2; ++mi)
#pragma unroll
        for (int ni = 0; ni < 2; ++ni) {
            const int col = n0 + wn * 32 + ni * 16 + l15;
            const float bv = bias[col];
#pragma unroll
            for (int r = 0; r < 8; ++r) {
                const int row = m0 + wm * 32 + mi * 16 + r + 8 * hi;
                const float val = acc[mi][ni][r] + bv;
                if constexpr (OUTF32) Of[(size_t)row * N + col] = val;
                else                  Ob[(size_t)row * N + col] = (__bf16)val;
            }
        }
}

// ---------------------------------------------------------------------------
// Flash-attention kernel. 128 threads = 4 waves; each wave owns 16 query rows
// (block: 64 query rows of one (b,h)). Streams key/value blocks of 32.
// Q and K tiles staged with async copies; V is transposed through VGPRs.
// ---------------------------------------------------------------------------
__global__ __launch_bounds__(128) void attn_wmma_kernel(const __bf16* __restrict__ qp,
                                                        const __bf16* __restrict__ kp,
                                                        const __bf16* __restrict__ vp,
                                                        __bf16* __restrict__ ctx) {
    constexpr int S = 2048, D = 1024, Hd = 64, H = 16;
    constexpr float SCALE = 0.125f;  // 1/sqrt(64)

    __shared__ __attribute__((aligned(16))) __bf16 Qs[64 * 72];     // [qrow][d], pad 8
    __shared__ __attribute__((aligned(16))) __bf16 Ks[32 * 72];     // [key][d],  pad 8
    __shared__ __attribute__((aligned(16))) __bf16 Vs[64 * 40];     // [d][key],  pad 8
    __shared__ __attribute__((aligned(16))) __bf16 Ps[4][16 * 40];  // per-wave P [row][key]

    const int t    = threadIdx.x;
    const int lane = t & 31;
    const int wave = t >> 5;
    const int hi   = lane >> 4;
    const int l15  = lane & 15;

    const int nq   = S / 64;  // 32 q-blocks per (b,h)
    const int sblk = blockIdx.x % nq;
    const int bh   = blockIdx.x / nq;
    const int h    = bh % H;
    const int b    = bh / H;

    const size_t base = ((size_t)b * S) * D + (size_t)h * Hd;
    const int q0 = sblk * 64;

    // --- stage Q block [64][64] via async copies ---
    for (int g = t; g < 512; g += 128) {
        const int row = g >> 3, c = (g & 7) * 8;
        async_b128(lds_off(&Qs[row * 72 + c]),
                   (uint32_t)((base + (size_t)(q0 + row) * D + c) * 2),
                   (uint64_t)qp);
    }
    wait_async0();
    __syncthreads();

    // --- Q fragments (held for whole kernel): A-matrix layout 16x32, two d-steps ---
    bf16x16 aq[2];
    {
        const int row = wave * 16 + l15;
#pragma unroll
        for (int kt = 0; kt < 2; ++kt) {
            bf16x8 lo = *(const bf16x8*)&Qs[row * 72 + kt * 32 + 8 * hi];
            bf16x8 h8 = *(const bf16x8*)&Qs[row * 72 + kt * 32 + 16 + 8 * hi];
#pragma unroll
            for (int i = 0; i < 8; ++i) { aq[kt][i] = lo[i]; aq[kt][i + 8] = h8[i]; }
        }
    }

    float m[8], l[8];
    f32x8 acc[4];
#pragma unroll
    for (int r = 0; r < 8; ++r) { m[r] = -1e30f; l[r] = 0.0f; }
#pragma unroll
    for (int dt = 0; dt < 4; ++dt)
#pragma unroll
        for (int r = 0; r < 8; ++r) acc[dt][r] = 0.0f;

    for (int kb = 0; kb < S; kb += 32) {
        __syncthreads();  // previous iteration's LDS reads complete
        // --- stage K block [32 keys][64 d] via async copies ---
        for (int g = t; g < 256; g += 128) {
            const int row = g >> 3, c = (g & 7) * 8;
            async_b128(lds_off(&Ks[row * 72 + c]),
                       (uint32_t)((base + (size_t)(kb + row) * D + c) * 2),
                       (uint64_t)kp);
        }
        // --- stage V block transposed -> [d][key] (through VGPRs) ---
        for (int g = t; g < 256; g += 128) {
            const int row = g >> 3, c = (g & 7) * 8;
            bf16x8 v8 = *(const bf16x8*)(vp + base + (size_t)(kb + row) * D + c);
#pragma unroll
            for (int j = 0; j < 8; ++j) Vs[(c + j) * 40 + row] = v8[j];
        }
        wait_async0();
        __syncthreads();

        // --- scores: S[16q x 32k] = q @ k^T, two 16-key tiles, two d-steps each ---
        f32x8 sc[2];
#pragma unroll
        for (int nt = 0; nt < 2; ++nt)
#pragma unroll
            for (int r = 0; r < 8; ++r) sc[nt][r] = 0.0f;

#pragma unroll
        for (int nt = 0; nt < 2; ++nt) {
            const int key = nt * 16 + l15;
#pragma unroll
            for (int kt = 0; kt < 2; ++kt) {
                bf16x16 bk;
                bf16x8 lo = *(const bf16x8*)&Ks[key * 72 + kt * 32 + 8 * hi];
                bf16x8 h8 = *(const bf16x8*)&Ks[key * 72 + kt * 32 + 16 + 8 * hi];
#pragma unroll
                for (int i = 0; i < 8; ++i) { bk[i] = lo[i]; bk[i + 8] = h8[i]; }
                sc[nt] = __builtin_amdgcn_wmma_f32_16x16x32_bf16(
                    false, aq[kt], false, bk, (short)0, sc[nt], false, false);
            }
        }
#pragma unroll
        for (int nt = 0; nt < 2; ++nt)
#pragma unroll
            for (int r = 0; r < 8; ++r) sc[nt][r] *= SCALE;

        // --- online softmax (row r+8*hi lives in VGPR r of this half-wave) ---
        float mnew[8], alpha[8];
#pragma unroll
        for (int r = 0; r < 8; ++r) {
            float v = fmaxf(sc[0][r], sc[1][r]);
            v = fmaxf(v, __shfl_xor(v, 1, 32));
            v = fmaxf(v, __shfl_xor(v, 2, 32));
            v = fmaxf(v, __shfl_xor(v, 4, 32));
            v = fmaxf(v, __shfl_xor(v, 8, 32));  // butterfly within 16-lane half
            mnew[r]  = fmaxf(m[r], v);
            alpha[r] = __expf(m[r] - mnew[r]);
        }
#pragma unroll
        for (int nt = 0; nt < 2; ++nt)
#pragma unroll
            for (int r = 0; r < 8; ++r) sc[nt][r] = __expf(sc[nt][r] - mnew[r]);
#pragma unroll
        for (int r = 0; r < 8; ++r) {
            float s = sc[0][r] + sc[1][r];
            s += __shfl_xor(s, 1, 32);
            s += __shfl_xor(s, 2, 32);
            s += __shfl_xor(s, 4, 32);
            s += __shfl_xor(s, 8, 32);
            l[r] = l[r] * alpha[r] + s;
            m[r] = mnew[r];
        }
#pragma unroll
        for (int dt = 0; dt < 4; ++dt)
#pragma unroll
            for (int r = 0; r < 8; ++r) acc[dt][r] *= alpha[r];

        // --- P (f32, C-layout) -> bf16 A-layout via per-wave LDS scratch ---
        __bf16* pw = &Ps[wave][0];
#pragma unroll
        for (int r = 0; r < 8; ++r) {
            const int row = r + 8 * hi;
            pw[row * 40 + l15]      = (__bf16)sc[0][r];
            pw[row * 40 + 16 + l15] = (__bf16)sc[1][r];
        }
        asm volatile("s_wait_dscnt 0" ::: "memory");  // wave-local LDS RAW fence

        bf16x16 pa;
        {
            const int row = l15;
            bf16x8 lo = *(const bf16x8*)&pw[row * 40 + 8 * hi];
            bf16x8 h8 = *(const bf16x8*)&pw[row * 40 + 16 + 8 * hi];
#pragma unroll
            for (int i = 0; i < 8; ++i) { pa[i] = lo[i]; pa[i + 8] = h8[i]; }
        }
        // --- acc += P @ V (4 d-tiles of 16) ---
#pragma unroll
        for (int dt = 0; dt < 4; ++dt) {
            const int d = dt * 16 + l15;
            bf16x16 vb;
            bf16x8 lo = *(const bf16x8*)&Vs[d * 40 + 8 * hi];
            bf16x8 h8 = *(const bf16x8*)&Vs[d * 40 + 16 + 8 * hi];
#pragma unroll
            for (int i = 0; i < 8; ++i) { vb[i] = lo[i]; vb[i + 8] = h8[i]; }
            acc[dt] = __builtin_amdgcn_wmma_f32_16x16x32_bf16(
                false, pa, false, vb, (short)0, acc[dt], false, false);
        }
    }

    // --- normalize and write context (bf16, [B*S, D] with head offset) ---
#pragma unroll
    for (int dt = 0; dt < 4; ++dt)
#pragma unroll
        for (int r = 0; r < 8; ++r) {
            const int row = q0 + wave * 16 + r + 8 * hi;
            const float o = acc[dt][r] / l[r];
            ctx[base + (size_t)row * D + dt * 16 + l15] = (__bf16)o;
        }
}

// ---------------------------------------------------------------------------
// Host-side launch
// ---------------------------------------------------------------------------
extern "C" void kernel_launch(void* const* d_in, const int* in_sizes, int n_in,
                              void* d_out, int out_size, void* d_ws, size_t ws_size,
                              hipStream_t stream) {
    (void)in_sizes; (void)n_in; (void)out_size; (void)ws_size;
    constexpr int Bsz = 4, S = 2048, D = 1024;
    constexpr int M = Bsz * S;  // 8192

    const float* Q   = (const float*)d_in[0];
    const float* Kin = (const float*)d_in[1];
    const float* V   = (const float*)d_in[2];
    const float* q_w = (const float*)d_in[3];
    const float* q_b = (const float*)d_in[4];
    const float* k_w = (const float*)d_in[5];
    const float* k_b = (const float*)d_in[6];
    const float* v_w = (const float*)d_in[7];
    const float* v_b = (const float*)d_in[8];
    const float* o_w = (const float*)d_in[9];
    const float* o_b = (const float*)d_in[10];
    float* out = (float*)d_out;

    // workspace layout (bf16 elements): 4 weights (1M each) + 4 activations (8M each)
    __bf16* wq  = (__bf16*)d_ws;
    __bf16* wk  = wq + (size_t)(1 << 20);
    __bf16* wv  = wk + (size_t)(1 << 20);
    __bf16* wo  = wv + (size_t)(1 << 20);
    __bf16* qp  = wo + (size_t)(1 << 20);
    __bf16* kp  = qp + (size_t)(8 << 20);
    __bf16* vp  = kp + (size_t)(8 << 20);
    __bf16* ctx = vp + (size_t)(8 << 20);

    const int nw = 1 << 20;  // D*D
    cvt_f32_bf16_kernel<<<nw / 1024, 256, 0, stream>>>(q_w, wq, nw);
    cvt_f32_bf16_kernel<<<nw / 1024, 256, 0, stream>>>(k_w, wk, nw);
    cvt_f32_bf16_kernel<<<nw / 1024, 256, 0, stream>>>(v_w, wv, nw);
    cvt_f32_bf16_kernel<<<nw / 1024, 256, 0, stream>>>(o_w, wo, nw);

    dim3 ggrid(D / 64, M / 128);  // (16, 64)
    gemm_wmma_kernel<true,  false><<<ggrid, 256, 0, stream>>>(Q,   wq, q_b, qp,  M, D, D);
    gemm_wmma_kernel<true,  false><<<ggrid, 256, 0, stream>>>(Kin, wk, k_b, kp,  M, D, D);
    gemm_wmma_kernel<true,  false><<<ggrid, 256, 0, stream>>>(V,   wv, v_b, vp,  M, D, D);

    attn_wmma_kernel<<<Bsz * 16 * (S / 64), 128, 0, stream>>>(qp, kp, vp, ctx);

    gemm_wmma_kernel<false, true><<<ggrid, 256, 0, stream>>>(ctx, wo, o_b, out, M, D, D);
}
